// VectorQuantizer_9706626089489
// MI455X (gfx1250) — compile-verified, matrix operand order checked by reading, not run on hardware
//
#include <hip/hip_runtime.h>
#include <hip/hip_bf16.h>

typedef __attribute__((ext_vector_type(16))) __bf16 v16bf;
typedef __attribute__((ext_vector_type(8)))  float  v8f;

#define C_DIM 256
#define E_DIM 1024
#define ROWS_PER_BLK 16

// ---------- helpers ----------
static __device__ inline uint4 ld_u4(const unsigned short* p) {
    return *(const uint4*)p;   // 16B aligned by construction -> ds_load_b128 / global_load_b128
}

static __device__ inline v16bf mk_frag(uint4 c0, uint4 c1) {
    union { uint4 q[2]; v16bf v; } u;
    u.q[0] = c0; u.q[1] = c1;
    return u.v;
}

// split f32 -> (hi, lo) bf16 by truncation; hi + lo reproduces x to ~2^-17 rel.
static __device__ inline void split_bf16(float x, unsigned short& hi, unsigned short& lo) {
    unsigned int b = __float_as_uint(x);
    hi = (unsigned short)(b >> 16);
    float hf = __uint_as_float(b & 0xFFFF0000u);
    float lf = x - hf;
    lo = (unsigned short)(__float_as_uint(lf) >> 16);
}

// ---------- kernel 1: codebook prep (split + norms) ----------
__global__ __launch_bounds__(256)
void vq_prep_emb(const float* __restrict__ emb,
                 unsigned short* __restrict__ ebh,
                 unsigned short* __restrict__ ebl,
                 float* __restrict__ enorm) {
    __shared__ float red[256];
    const int e = blockIdx.x;        // embedding row
    const int c = threadIdx.x;       // channel
    const int off = e * C_DIM + c;
    float x = emb[off];
    unsigned short h, l;
    split_bf16(x, h, l);
    ebh[off] = h;
    ebl[off] = l;
    red[c] = x * x;
    __syncthreads();
    #pragma unroll
    for (int s = 128; s > 0; s >>= 1) {
        if (c < s) red[c] += red[c + s];
        __syncthreads();
    }
    if (c == 0) enorm[e] = red[0];
}

// ---------- kernel 2: distance GEMM (split-bf16 WMMA) + argmin ----------
__global__ __launch_bounds__(256)
void vq_argmin(const float* __restrict__ z,
               const unsigned short* __restrict__ ebh,
               const unsigned short* __restrict__ ebl,
               const float* __restrict__ enorm,
               int*   __restrict__ idx_out,
               float* __restrict__ idxf_out) {
    __shared__ __align__(16) unsigned short zh[ROWS_PER_BLK * C_DIM]; // 8 KB
    __shared__ __align__(16) unsigned short zl[ROWS_PER_BLK * C_DIM]; // 8 KB
    __shared__ float cval[8][ROWS_PER_BLK];
    __shared__ int   cidx[8][ROWS_PER_BLK];

    const int t    = threadIdx.x;
    const int blk  = blockIdx.x;        // 16-row strip of z_flat
    const int lane = t & 31;
    const int wave = t >> 5;            // 8 waves per block
    const int half = lane >> 4;         // 0: lanes 0-15, 1: lanes 16-31
    const int l16  = lane & 15;

    // ---- stage z strip to LDS as bf16 hi/lo (coalesced f32 reads) ----
    const float* zg = z + (size_t)blk * (ROWS_PER_BLK * C_DIM);
    #pragma unroll
    for (int i = 0; i < 16; ++i) {
        int off = i * C_DIM + t;        // 256 threads cover one row per i
        unsigned short h, l;
        split_bf16(zg[off], h, l);
        zh[off] = h;
        zl[off] = l;
    }
    __syncthreads();

    // A-fragment base for this lane: row M = l16, K offset per half (ISA 7.12.2)
    const int abase = l16 * C_DIM + half * 8;

    float minv[8];
    int   mini[8];
    #pragma unroll
    for (int v = 0; v < 8; ++v) { minv[v] = __builtin_inff(); mini[v] = 0; }

    // each wave owns 8 of the 64 column tiles (ascending -> argmin tie keeps lowest idx)
    for (int tt = 0; tt < 8; ++tt) {
        const int et = wave * 8 + tt;
        const int e0 = et * 16;
        const unsigned short* bhp = ebh + (size_t)(e0 + l16) * C_DIM + half * 8;
        const unsigned short* blp = ebl + (size_t)(e0 + l16) * C_DIM + half * 8;

        v8f acc0 = {0.f, 0.f, 0.f, 0.f, 0.f, 0.f, 0.f, 0.f};
        v8f acc1 = {0.f, 0.f, 0.f, 0.f, 0.f, 0.f, 0.f, 0.f};
        v8f acc2 = {0.f, 0.f, 0.f, 0.f, 0.f, 0.f, 0.f, 0.f};

        #pragma unroll
        for (int kc = 0; kc < 8; ++kc) {
            const int kb = kc * 32;
            v16bf ah = mk_frag(ld_u4(zh + abase + kb), ld_u4(zh + abase + kb + 16));
            v16bf al = mk_frag(ld_u4(zl + abase + kb), ld_u4(zl + abase + kb + 16));
            v16bf bh = mk_frag(ld_u4(bhp + kb),        ld_u4(bhp + kb + 16));
            v16bf bl = mk_frag(ld_u4(blp + kb),        ld_u4(blp + kb + 16));
            // fp32 emulated via 3 bf16 products (alo*blo term ~2^-18, dropped)
            acc0 = __builtin_amdgcn_wmma_f32_16x16x32_bf16(false, ah, false, bh, (short)0, acc0, false, false);
            acc1 = __builtin_amdgcn_wmma_f32_16x16x32_bf16(false, ah, false, bl, (short)0, acc1, false, false);
            acc2 = __builtin_amdgcn_wmma_f32_16x16x32_bf16(false, al, false, bh, (short)0, acc2, false, false);
        }

        const int   n  = e0 + l16;          // this lane's column (embedding id)
        const float en = enorm[n];
        #pragma unroll
        for (int v = 0; v < 8; ++v) {
            float dot = acc0[v] + acc1[v] + acc2[v];
            float d = en - 2.0f * dot;      // ||z||^2 constant per row: dropped
            if (d < minv[v]) { minv[v] = d; mini[v] = n; }
        }
    }

    // ---- intra-wave reduction: min over the 16 lanes of each half ----
    // C layout: VGPR v -> row (v + 8*half), column = l16 (+16*et). xor masks <16 keep halves separate.
    #pragma unroll
    for (int v = 0; v < 8; ++v) {
        float val = minv[v];
        int   id  = mini[v];
        #pragma unroll
        for (int m = 1; m < 16; m <<= 1) {
            float ov = __shfl_xor(val, m, 32);
            int   oi = __shfl_xor(id,  m, 32);
            if (ov < val || (ov == val && oi < id)) { val = ov; id = oi; }
        }
        if (l16 == 0) {                       // lane 0 -> rows 0..7, lane 16 -> rows 8..15
            cval[wave][half * 8 + v] = val;
            cidx[wave][half * 8 + v] = id;
        }
    }
    __syncthreads();

    // ---- cross-wave reduction over the 8 column-tile groups ----
    if (t < ROWS_PER_BLK) {
        float best = cval[0][t];
        int   bid  = cidx[0][t];
        #pragma unroll
        for (int w = 1; w < 8; ++w) {
            float v2 = cval[w][t];
            int   i2 = cidx[w][t];
            if (v2 < best || (v2 == best && i2 < bid)) { best = v2; bid = i2; }
        }
        const int row = blk * ROWS_PER_BLK + t;
        idx_out[row]  = bid;
        idxf_out[row] = (float)bid;           // indices output (flat == (B, H*W) order)
    }
}

// ---------- kernel 3: gather z_q = emb[idx] ----------
__global__ __launch_bounds__(256)
void vq_gather(const float* __restrict__ emb,
               const int* __restrict__ idx,
               float* __restrict__ zq) {
    const int t   = threadIdx.x;
    const int row = blockIdx.x * 4 + (t >> 6);   // 4 rows per block
    const int c4  = (t & 63) << 2;               // float4 per thread
    const int e   = idx[row];
    const float4 v = *(const float4*)(emb + (size_t)e * C_DIM + c4);
    *(float4*)(zq + (size_t)row * C_DIM + c4) = v;
}

// ---------- launcher ----------
extern "C" void kernel_launch(void* const* d_in, const int* in_sizes, int n_in,
                              void* d_out, int out_size, void* d_ws, size_t ws_size,
                              hipStream_t stream) {
    const float* z   = (const float*)d_in[0];   // (32,32,32,256) f32 -> N=32768 rows
    const float* emb = (const float*)d_in[1];   // (1024,256) f32

    const int N = in_sizes[0] / C_DIM;          // 32768
    const int E = in_sizes[1] / C_DIM;          // 1024

    // workspace layout (16B-aligned sections)
    char* ws = (char*)d_ws;
    unsigned short* ebh   = (unsigned short*)(ws);                         // 512 KB
    unsigned short* ebl   = (unsigned short*)(ws + (size_t)E * C_DIM * 2); // 512 KB
    float*          enorm = (float*)(ws + (size_t)E * C_DIM * 4);          // 4 KB
    int*            idxi  = (int*)  (ws + (size_t)E * C_DIM * 4 + 4096);   // 128 KB

    float* zq   = (float*)d_out;                // N*C floats
    float* idxf = (float*)d_out + (size_t)N * C_DIM;

    vq_prep_emb<<<E, 256, 0, stream>>>(emb, ebh, ebl, enorm);
    vq_argmin <<<N / ROWS_PER_BLK, 256, 0, stream>>>(z, ebh, ebl, enorm, idxi, idxf);
    vq_gather <<<N / 4, 256, 0, stream>>>(emb, idxi, zq);
}
